// MeshEncoder_52458730553616
// MI455X (gfx1250) — matrix operand order repaired and checked by reading, not compiled, not run on hardware
//
#include <hip/hip_runtime.h>
#include <stdint.h>

#define B_   8
#define V_   2562
#define F_   5120
#define NL   17
#define VP   2624        // ceil64(V): M padded so grid.x*64 tiles exactly cover it
#define P_   320         // activation/weight pitch: multiple of 32, >= 300
#define WW   81          // u32 words per adjacency bit-row (ceil(2562/32))
#define MAXN 512         // neighbor-list capacity per vertex

#if defined(__has_builtin)
#if __has_builtin(__builtin_amdgcn_global_load_async_to_lds_b128) && \
    __has_builtin(__builtin_amdgcn_s_wait_asynccnt)
#define USE_ASYNC_LDS 1
#endif
#endif

typedef __attribute__((ext_vector_type(16))) __bf16          bf16x16;
typedef __attribute__((ext_vector_type(8)))  float           f32x8;
typedef __attribute__((ext_vector_type(8)))  unsigned short  u16x8;

#ifdef USE_ASYNC_LDS
typedef __attribute__((ext_vector_type(4))) int              v4i_;
typedef __attribute__((address_space(1)))   v4i_*            as1_v4i;
typedef __attribute__((address_space(3)))   v4i_*            as3_v4i;
#endif

__device__ __forceinline__ unsigned short f2bf(float x) {
    unsigned u = __builtin_bit_cast(unsigned, x);
    u += 0x7FFFu + ((u >> 16) & 1u);               // round-to-nearest-even
    return (unsigned short)(u >> 16);
}
__device__ __forceinline__ float bf2f(unsigned short h) {
    unsigned u = ((unsigned)h) << 16;
    return __builtin_bit_cast(float, u);
}

// ---- adjacency: set diagonal bits (buffer pre-zeroed) ----
__global__ void diag_kernel(unsigned* __restrict__ adjb) {
    int idx = blockIdx.x * blockDim.x + threadIdx.x;
    if (idx >= B_ * V_) return;
    int v = idx % V_;
    adjb[(size_t)idx * WW + (v >> 5)] |= (1u << (v & 31));
}

// ---- adjacency: OR in the 6 directed edges of each face ----
__global__ void edge_kernel(const int* __restrict__ faces, unsigned* __restrict__ adjb) {
    int idx = blockIdx.x * blockDim.x + threadIdx.x;
    if (idx >= B_ * F_) return;
    int b = idx / F_;
    const int* fc = faces + (size_t)idx * 3;
    int v1 = fc[0], v2 = fc[1], v3 = fc[2];
    unsigned* base = adjb + (size_t)b * V_ * WW;
    auto set1 = [&](int r, int c) {
        atomicOr(base + (size_t)r * WW + (c >> 5), 1u << (c & 31));
    };
    set1(v1, v2); set1(v1, v3);
    set1(v2, v1); set1(v2, v3);
    set1(v3, v1); set1(v3, v2);
}

// ---- initial activations: verts (V x 3) -> bf16, zero-padded buffer ----
__global__ void vin_kernel(const float* __restrict__ verts, unsigned short* __restrict__ A) {
    int idx = blockIdx.x * blockDim.x + threadIdx.x;
    if (idx >= B_ * V_ * 3) return;
    int k  = idx % 3;
    int bv = idx / 3;
    int v  = bv % V_;
    int b  = bv / V_;
    A[(size_t)b * VP * P_ + (size_t)v * P_ + k] = f2bf(verts[idx]);
}

// ---- weights: f32 (din x dout) -> bf16 padded 320x320 slot (pre-zeroed) ----
__global__ void wfill_kernel(const float* __restrict__ Wsrc, unsigned short* __restrict__ Wdst,
                             int din, int dout) {
    int idx = blockIdx.x * blockDim.x + threadIdx.x;
    if (idx >= din * dout) return;
    int r = idx / dout, c = idx % dout;
    Wdst[(size_t)r * P_ + c] = f2bf(Wsrc[idx]);
}

// ---- dense GEMM  XW = A(bf16) x W(bf16), f32 accumulate via WMMA ----
// Block = 4 waves. Block tile = 64(M) x 64(N). Wave w owns M-tile (bx*64+w*16),
// computes 16x64 = 4 accumulators, reusing one A fragment per K-step.
// B K-slab (32x64 bf16 = 4KB) is staged in double-buffered LDS, shared by all
// 4 waves; staged with GLOBAL_LOAD_ASYNC_TO_LDS_B128 (ASYNCcnt) when available.
// Fragment layouts per CDNA5 ISA 7.12.2:
//   A: lane m=l&15, hi=l>>4: halves[0..7]=A[m][k0+hi*8..], halves[8..15]=A[m][k0+16+hi*8..]
//   B: lane n=l&15, hi=l>>4: halves[0..7]=B[k0+hi*8..][n], halves[8..15]=B[k0+16+hi*8..][n]
//   D: vgpr r, lane l<16 -> (M=r, N=l); lane>=16 -> (M=8+r, N=l-16)
__global__ void __launch_bounds__(128) gemm_kernel(const unsigned short* __restrict__ Ag,
                                                   const unsigned short* __restrict__ Wb,
                                                   float* __restrict__ XW, int dinp) {
    __shared__ __align__(16) unsigned short Bs[2][32 * 64];

    const int tid  = threadIdx.x;
    const int lane = tid & 31;
    const int wv   = tid >> 5;
    const int n0   = blockIdx.y * 64;
    const unsigned short* A = Ag + (size_t)blockIdx.z * VP * P_;
    float*                Y = XW + (size_t)blockIdx.z * VP * P_;

    const int m  = lane & 15;
    const int hi = lane >> 4;
    const unsigned short* arow = A + (size_t)(blockIdx.x * 64 + wv * 16 + m) * P_;

    // B-staging assignment: 256 16-byte segments (32 rows x 8), 2 per thread.
    const int s0 = tid, s1 = tid + 128;
    const int r0 = s0 >> 3, p0 = s0 & 7;
    const int r1 = s1 >> 3, p1 = s1 & 7;

    f32x8 acc[4] = {};
    union Frag { u16x8 h[2]; bf16x16 v; };

    for (int k0 = 0, it = 0; k0 < dinp; k0 += 32, ++it) {
        const int buf = it & 1;
        const unsigned short* g0 = Wb + (size_t)(k0 + r0) * P_ + n0 + p0 * 8;
        const unsigned short* g1 = Wb + (size_t)(k0 + r1) * P_ + n0 + p1 * 8;
        unsigned short* l0 = &Bs[buf][r0 * 64 + p0 * 8];
        unsigned short* l1 = &Bs[buf][r1 * 64 + p1 * 8];
#ifdef USE_ASYNC_LDS
        __builtin_amdgcn_global_load_async_to_lds_b128(
            (as1_v4i)(void*)g0, (as3_v4i)(void*)l0, 0, 0);
        __builtin_amdgcn_global_load_async_to_lds_b128(
            (as1_v4i)(void*)g1, (as3_v4i)(void*)l1, 0, 0);
        __builtin_amdgcn_s_wait_asynccnt(0);
#else
        *(u16x8*)l0 = *(const u16x8*)g0;
        *(u16x8*)l1 = *(const u16x8*)g1;
#endif
        __syncthreads();

        Frag af;
        af.h[0] = *(const u16x8*)(arow + k0 + hi * 8);
        af.h[1] = *(const u16x8*)(arow + k0 + 16 + hi * 8);
        if (k0 + 32 < dinp) __builtin_prefetch(arow + k0 + 32, 0, 0);
#pragma unroll
        for (int j = 0; j < 4; ++j) {
            Frag bfr;
            for (int s = 0; s < 8; ++s) bfr.h[0][s] = Bs[buf][(hi * 8 + s) * 64 + j * 16 + m];
            for (int s = 0; s < 8; ++s) bfr.h[1][s] = Bs[buf][(16 + hi * 8 + s) * 64 + j * 16 + m];
            acc[j] = __builtin_amdgcn_wmma_f32_16x16x32_bf16(
                /*neg_a=*/false, af.v, /*neg_b=*/false, bfr.v,
                /*c_mod=*/(short)0, acc[j], /*reuse_a=*/false, /*reuse_b=*/false);
        }
    }

    const int n  = lane & 15;
    const int mb = blockIdx.x * 64 + wv * 16 + (hi ? 8 : 0);
#pragma unroll
    for (int j = 0; j < 4; ++j) {
        float* yp = Y + (size_t)mb * P_ + n0 + j * 16 + n;
#pragma unroll
        for (int r = 0; r < 8; ++r) yp[(size_t)r * P_] = acc[j][r];
    }
}

// ---- sparse neighborhood average + bias (+ elu), bf16 output for next layer ----
__global__ void agg_kernel(const unsigned* __restrict__ adjb, const float* __restrict__ XW,
                           const float* __restrict__ bias, unsigned short* __restrict__ Aout,
                           int dout, int apply_elu) {
    __shared__ unsigned nbr[MAXN];
    __shared__ int cnt;
    __shared__ int deg;
    const int b = blockIdx.x / V_;
    const int v = blockIdx.x % V_;
    const unsigned* row = adjb + ((size_t)b * V_ + v) * WW;
    if (threadIdx.x == 0) { cnt = 0; deg = 0; }
    __syncthreads();
    for (int w = threadIdx.x; w < WW; w += blockDim.x) {
        unsigned bits = row[w];
        if (bits) {
            int nb = __popc(bits);
            atomicAdd(&deg, nb);
            int pos = atomicAdd(&cnt, nb);
            while (bits) {
                int t = __ffs(bits) - 1;
                bits &= bits - 1;
                if (pos < MAXN) nbr[pos] = (unsigned)(w * 32 + t);
                ++pos;
            }
        }
    }
    __syncthreads();
    const int   mcnt = cnt < MAXN ? cnt : MAXN;
    const float inv  = 1.0f / (float)deg;     // deg >= 1 (diagonal always set)
    const float* Xb = XW + (size_t)b * VP * P_;
    unsigned short* Ab = Aout + (size_t)b * VP * P_;
    for (int c = threadIdx.x; c < dout; c += blockDim.x) {
        float s = 0.f;
        for (int i = 0; i < mcnt; ++i) s += Xb[(size_t)nbr[i] * P_ + c];
        float y = s * inv + bias[c];
        if (apply_elu) y = (y > 0.f) ? y : (__expf(y) - 1.f);
        Ab[(size_t)v * P_ + c] = f2bf(y);
    }
}

// ---- final: per-batch column max over real vertices, then elu ----
__global__ void reduce_kernel(const unsigned short* __restrict__ A, float* __restrict__ out) {
    const int b = blockIdx.x;
    const int c = threadIdx.x;  // 128 latent channels
    const unsigned short* Ab = A + (size_t)b * VP * P_ + c;
    float mx = -3.4e38f;
    for (int v = 0; v < V_; ++v) {
        float x = bf2f(Ab[(size_t)v * P_]);
        mx = mx > x ? mx : x;
    }
    out[b * 128 + c] = (mx > 0.f) ? mx : (__expf(mx) - 1.f);
}

static const int DIN[NL]  = {3, 60, 60, 60, 60, 120, 120, 120, 150, 200, 210, 250, 300, 300, 300, 300, 300};
static const int DOUT[NL] = {60, 60, 60, 60, 120, 120, 120, 150, 200, 210, 250, 300, 300, 300, 300, 300, 128};

extern "C" void kernel_launch(void* const* d_in, const int* in_sizes, int n_in,
                              void* d_out, int out_size, void* d_ws, size_t ws_size,
                              hipStream_t stream) {
    const float* verts = (const float*)d_in[0];
    const int*   faces = (const int*)d_in[1];
    // dict order: verts, faces, weights[0..16], biases[0..16]

    char* ws = (char*)d_ws;
    size_t off = 0;
    auto wsalloc = [&](size_t bytes) -> void* {
        void* p = ws + off;
        off = (off + bytes + 255) & ~(size_t)255;
        return p;
    };
    unsigned*       adjb = (unsigned*)      wsalloc((size_t)B_ * V_ * WW * 4);   // ~6.6 MB
    unsigned short* A    = (unsigned short*)wsalloc((size_t)B_ * VP * P_ * 2);   // ~13.4 MB
    float*          XW   = (float*)         wsalloc((size_t)B_ * VP * P_ * 4);   // ~26.9 MB
    unsigned short* Wb   = (unsigned short*)wsalloc((size_t)NL * P_ * P_ * 2);   // ~3.5 MB

    (void)hipMemsetAsync(adjb, 0, (size_t)B_ * V_ * WW * 4, stream);
    (void)hipMemsetAsync(A,    0, (size_t)B_ * VP * P_ * 2, stream);
    (void)hipMemsetAsync(Wb,   0, (size_t)NL * P_ * P_ * 2, stream);

    diag_kernel<<<(B_ * V_ + 255) / 256, 256, 0, stream>>>(adjb);
    edge_kernel<<<(B_ * F_ + 255) / 256, 256, 0, stream>>>(faces, adjb);
    vin_kernel <<<(B_ * V_ * 3 + 255) / 256, 256, 0, stream>>>(verts, A);
    for (int l = 0; l < NL; ++l) {
        int n = DIN[l] * DOUT[l];
        wfill_kernel<<<(n + 255) / 256, 256, 0, stream>>>(
            (const float*)d_in[2 + l], Wb + (size_t)l * P_ * P_, DIN[l], DOUT[l]);
    }

    for (int l = 0; l < NL; ++l) {
        int dinp  = (DIN[l]  + 31) & ~31;   // K padded to 32
        int doutp = (DOUT[l] + 15) & ~15;   // N padded to 16
        dim3 g(VP / 64, (doutp + 63) / 64, B_);
        gemm_kernel<<<g, 128, 0, stream>>>(A, Wb + (size_t)l * P_ * P_, XW, dinp);
        agg_kernel<<<B_ * V_, 256, 0, stream>>>(
            adjb, XW, (const float*)d_in[2 + NL + l], A, DOUT[l], (l < NL - 1) ? 1 : 0);
    }

    reduce_kernel<<<B_, 128, 0, stream>>>(A, (float*)d_out);
}
